// TransformerLayer_37486474559633
// MI455X (gfx1250) — compile-verified, matrix operand order checked by reading, not run on hardware
//
#include <hip/hip_runtime.h>

// ---------------------------------------------------------------------------
// Types
// ---------------------------------------------------------------------------
typedef __bf16 bf16;
typedef __attribute__((ext_vector_type(16))) bf16  bf16x16;
typedef __attribute__((ext_vector_type(8)))  bf16  bf16x8;
typedef __attribute__((ext_vector_type(8)))  float f32x8;
typedef __attribute__((ext_vector_type(4)))  float f32x4;

constexpr int DMODEL = 512;
constexpr int HEADS  = 8;
constexpr int HD     = 64;
constexpr int DFF    = 2048;
constexpr int MAXN   = 1024;   // max sequence length in this workload

// ---------------------------------------------------------------------------
// WMMA helpers (CDNA5 wave32, 16x16x32 bf16 -> f32)
// ---------------------------------------------------------------------------
__device__ __forceinline__ f32x8 wmma_bf16(bf16x16 a, bf16x16 b, f32x8 c) {
  return __builtin_amdgcn_wmma_f32_16x16x32_bf16(
      /*neg_a=*/false, a, /*neg_b=*/false, b,
      /*c_mod=*/(short)0, c, /*reuse_a=*/false, /*reuse_b=*/false);
}

// Fragment load: per ISA layout, each lane's 16 bf16 live in two groups of 8
// contiguous K values: [lane_hi8, lane_hi8+8) and [16+lane_hi8, 16+lane_hi8+8)
// where lane_hi8 = (lane>>4)*8.  row points at the K=0 element of this lane's
// row; rows must be 16B-aligned with 16B-multiple stride.
__device__ __forceinline__ bf16x16 load_frag(const bf16* row, int lane_hi8) {
  union { bf16x16 v; bf16x8 h[2]; } u;
  u.h[0] = *(const bf16x8*)(row + lane_hi8);
  u.h[1] = *(const bf16x8*)(row + 16 + lane_hi8);
  return u.v;
}

__device__ __forceinline__ float gelu_exact(float x) {
  return 0.5f * x * (1.0f + erff(x * 0.70710678118654752f));
}

// ---------------------------------------------------------------------------
// CDNA5 async global->LDS copy (ASYNCcnt-tracked, no VGPR data round trip).
// GV mode: per-lane 64-bit global address, per-lane 32-bit LDS byte offset.
// Generic LDS pointers carry the LDS byte offset in their low 32 bits.
// ---------------------------------------------------------------------------
__device__ __forceinline__ void async_copy_b128(unsigned lds_off, const void* gptr) {
  asm volatile("global_load_async_to_lds_b128 %0, %1, off"
               :: "v"(lds_off), "v"((unsigned long long)(size_t)gptr)
               : "memory");
}
__device__ __forceinline__ void wait_async0() {
  asm volatile("s_wait_asynccnt 0x0" ::: "memory");
}

// ---------------------------------------------------------------------------
// fp32 -> bf16 conversion
// ---------------------------------------------------------------------------
__global__ __launch_bounds__(256)
void cvt_f32_to_bf16_kernel(const float* __restrict__ src, bf16* __restrict__ dst, int n) {
  const int i = (blockIdx.x * 256 + threadIdx.x) * 4;
  if (i + 4 <= n) {
    f32x4 v = *(const f32x4*)(src + i);
    dst[i + 0] = (bf16)v[0];
    dst[i + 1] = (bf16)v[1];
    dst[i + 2] = (bf16)v[2];
    dst[i + 3] = (bf16)v[3];
  } else {
    for (int j = i; j < n; ++j) dst[j] = (bf16)src[j];
  }
}

// ---------------------------------------------------------------------------
// Tiled GEMM: C[M x N] = A[M x K] @ B[K x N] + bias, bf16 in, f32 accum.
// Block = 128 threads (4 waves). Tile 128x64, K-step 32, double-buffered LDS.
// A tile staged with GLOBAL_LOAD_ASYNC_TO_LDS_B128 (one row per thread);
// B tile staged transposed ([n][k]) through registers for fragment loads.
// Wave w owns rows [w*32, w*32+32): 2 A-frags x 4 B-frags = 8 WMMA / k-step,
// each B fragment reused by both A fragments.
// EPI: 0 = f32 out, 1 = bf16 out, 2 = GELU -> bf16 out. Bias always applied.
// ---------------------------------------------------------------------------
constexpr int BM = 128, BN = 64, BK = 32, BKP = BK + 8;  // 40 elems = 80B rows

template<int EPI>
__global__ __launch_bounds__(128)
void gemm_bf16_kernel(const bf16* __restrict__ A, int lda,
                      const bf16* __restrict__ B, int ldb,
                      const float* __restrict__ bias,
                      float* __restrict__ Cf, bf16* __restrict__ Cb, int ldc,
                      int M, int K) {
  __shared__ alignas(16) bf16 As[2][BM][BKP];
  __shared__ alignas(16) bf16 Bs[2][BN][BKP];   // stored transposed: [n][k]

  const int tid      = threadIdx.x;
  const int lane     = tid & 31;
  const int wave     = tid >> 5;
  const int lane_lo  = lane & 15;
  const int lane_hi8 = (lane >> 4) << 3;
  const size_t m0 = (size_t)blockIdx.y * BM;
  const size_t n0 = (size_t)blockIdx.x * BN;

  f32x8 acc[2][4] = {};

  // staging roles
  const int bk = tid >> 2;            // B k row 0..31
  const int bn = (tid & 3) << 4;      // B n quarter 0/16/32/48

  auto stage = [&](int buf, int k0) {
    {  // A: 128 rows x 32 k, one row per thread, 4x B128 async copies
      size_t arow = m0 + tid;
      if (arow >= (size_t)M) arow = (size_t)M - 1;
      const bf16* src = A + arow * (size_t)lda + k0;
      const unsigned dst = (unsigned)(size_t)(void*)&As[buf][tid][0];
      async_copy_b128(dst,      src);
      async_copy_b128(dst + 16, src + 8);
      async_copy_b128(dst + 32, src + 16);
      async_copy_b128(dst + 48, src + 24);
      __builtin_prefetch(src + BK, 0, 1);          // -> global_prefetch_b8
    }
    {  // B: 32 k x 64 n, transpose through registers into Bs[n][k]
      const bf16* src = B + (size_t)(k0 + bk) * ldb + n0 + bn;
      bf16 t[16];
      *(f32x4*)&t[0] = *(const f32x4*)src;
      *(f32x4*)&t[8] = *(const f32x4*)(src + 8);
      __builtin_prefetch(src + (size_t)BK * ldb, 0, 1);
      #pragma unroll
      for (int i = 0; i < 16; ++i) Bs[buf][bn + i][bk] = t[i];
    }
  };

  stage(0, 0);
  wait_async0();
  __syncthreads();
  const int nsteps = K / BK;
  for (int s = 0; s < nsteps; ++s) {
    const int buf = s & 1;
    if (s + 1 < nsteps) stage(buf ^ 1, (s + 1) * BK);
    bf16x16 a0 = load_frag(&As[buf][wave * 32 + lane_lo][0],      lane_hi8);
    bf16x16 a1 = load_frag(&As[buf][wave * 32 + 16 + lane_lo][0], lane_hi8);
    #pragma unroll
    for (int c = 0; c < 4; ++c) {
      bf16x16 bfrag = load_frag(&Bs[buf][c * 16 + lane_lo][0], lane_hi8);
      acc[0][c] = wmma_bf16(a0, bfrag, acc[0][c]);
      acc[1][c] = wmma_bf16(a1, bfrag, acc[1][c]);
    }
    wait_async0();           // async stage complete before barrier
    __syncthreads();
  }

  // Epilogue. C/D layout: VGPR r -> row (r + 8*(lane>>4)), col = lane&15.
  #pragma unroll
  for (int g = 0; g < 2; ++g) {
    const size_t mrow = m0 + (size_t)wave * 32 + g * 16;
    #pragma unroll
    for (int c = 0; c < 4; ++c) {
      const int col = (int)n0 + c * 16 + lane_lo;
      const float bv = bias[col];
      #pragma unroll
      for (int r = 0; r < 8; ++r) {
        const size_t row = mrow + r + ((lane >> 4) << 3);
        if (row < (size_t)M) {
          float val = acc[g][c][r] + bv;
          if (EPI == 2) val = gelu_exact(val);
          if (EPI == 0) Cf[row * (size_t)ldc + col] = val;
          else          Cb[row * (size_t)ldc + col] = (bf16)val;
        }
      }
    }
  }
}

// ---------------------------------------------------------------------------
// Attention: one wave (32 threads) per (seq, head, 16-query tile).
//  1) S = scale * Q Kt via WMMA (2 per 16-key tile, head dim 64) -> LDS f32
//  2) fp32 softmax in LDS; P stored as unnormalized exp in bf16; 1/sum folded
//     into PV epilogue per output row.
//  3) ctx = P @ V via WMMA, V staged transposed in LDS per 32-key chunk.
// ---------------------------------------------------------------------------
constexpr int    SROW    = MAXN + 8;                 // f32 score row stride
constexpr int    PROW    = MAXN + 8;                 // bf16 P row stride
constexpr int    VROW    = 40;                       // bf16 Vt row stride (80B)
constexpr size_t SBYTES  = (size_t)16 * SROW * 4;
constexpr size_t PBYTES  = (size_t)16 * PROW * 2;
constexpr size_t VTBYTES = (size_t)64 * VROW * 2;
constexpr size_t ATTN_SMEM = SBYTES + PBYTES + VTBYTES + 64 * 4;

__global__ __launch_bounds__(32)
void attn_kernel(const bf16* __restrict__ q, const bf16* __restrict__ k,
                 const bf16* __restrict__ v, const long long* __restrict__ cum,
                 bf16* __restrict__ ctx) {
  extern __shared__ char smem[];
  float* S   = (float*)smem;
  bf16*  P   = (bf16*)(smem + SBYTES);
  bf16*  Vt  = (bf16*)(smem + SBYTES + PBYTES);
  float* red = (float*)(smem + SBYTES + PBYTES + VTBYTES);

  const int bi = blockIdx.z, hh = blockIdx.y, qt = blockIdx.x;
  const long long s0 = cum[bi];
  const int len = (int)(cum[bi + 1] - s0);
  if (qt * 16 >= len) return;

  const int lane     = threadIdx.x;
  const int lane_lo  = lane & 15;
  const int half     = lane >> 4;
  const int lane_hi8 = half << 3;
  const float scale  = 0.125f;                 // 1/sqrt(64)

  int qidx = qt * 16 + lane_lo;
  if (qidx >= len) qidx = len - 1;             // clamp (lengths here are /16)
  const size_t qrow = (size_t)s0 + qidx;

  const bf16* qbase = q + qrow * DMODEL + hh * HD;
  bf16x16 qf0 = load_frag(qbase, lane_hi8);
  bf16x16 qf1 = load_frag(qbase + 32, lane_hi8);

  // ---- scores ----
  const int nkt = (len + 15) >> 4;
  for (int kt = 0; kt < nkt; ++kt) {
    const int key = kt * 16 + lane_lo;
    const size_t krow = (size_t)s0 + (key < len ? key : len - 1);
    const bf16* kbase = k + krow * DMODEL + hh * HD;
    f32x8 sc = {};
    sc = wmma_bf16(qf0, load_frag(kbase, lane_hi8), sc);
    sc = wmma_bf16(qf1, load_frag(kbase + 32, lane_hi8), sc);
    #pragma unroll
    for (int r = 0; r < 8; ++r) {
      const int row = r + (half << 3);
      S[row * SROW + key] = (key < len) ? sc[r] * scale : -3.0e38f;
    }
  }
  __syncthreads();

  // ---- softmax (two lanes per row: even/odd columns) ----
  const int row = lane_lo;
  float mx = -3.0e38f;
  for (int c = half; c < len; c += 2) mx = fmaxf(mx, S[row * SROW + c]);
  red[half * 16 + row] = mx;
  __syncthreads();
  mx = fmaxf(red[row], red[16 + row]);
  float sum = 0.f;
  for (int c = half; c < len; c += 2) {
    const float p = __expf(S[row * SROW + c] - mx);
    P[row * PROW + c] = (bf16)p;               // unnormalized
    sum += p;
  }
  red[half * 16 + row] = sum;
  __syncthreads();
  if (half == 0) red[32 + row] = 1.0f / (red[row] + red[16 + row]);
  const int lenr = ((len + 31) >> 5) << 5;     // pad keys to 32 for WMMA
  if (lane < 16)
    for (int c = len; c < lenr; ++c) P[lane * PROW + c] = (bf16)0.f;
  __syncthreads();

  // ---- ctx = P @ V ----
  f32x8 acc[4] = {};
  const int nkc = lenr >> 5;
  for (int c32 = 0; c32 < nkc; ++c32) {
    const int key = c32 * 32 + lane;           // each lane stages one V row
    bf16 t[64];
    if (key < len) {
      const bf16* vsrc = v + ((size_t)s0 + key) * DMODEL + hh * HD;
      #pragma unroll
      for (int i = 0; i < 8; ++i) *(f32x4*)&t[i * 8] = *(const f32x4*)(vsrc + i * 8);
    } else {
      #pragma unroll
      for (int i = 0; i < 64; ++i) t[i] = (bf16)0.f;
    }
    #pragma unroll
    for (int i = 0; i < 64; ++i) Vt[i * VROW + lane] = t[i];   // transpose
    __syncthreads();
    bf16x16 pf = load_frag(P + (size_t)lane_lo * PROW + c32 * 32, lane_hi8);
    #pragma unroll
    for (int c = 0; c < 4; ++c) {
      bf16x16 vf = load_frag(Vt + (size_t)(c * 16 + lane_lo) * VROW, lane_hi8);
      acc[c] = wmma_bf16(pf, vf, acc[c]);
    }
    __syncthreads();
  }

  #pragma unroll
  for (int c = 0; c < 4; ++c) {
    const int col = hh * HD + c * 16 + lane_lo;
    #pragma unroll
    for (int r = 0; r < 8; ++r) {
      const int rowi = r + (half << 3);
      if (qt * 16 + rowi < len) {
        const size_t grow = (size_t)s0 + qt * 16 + rowi;
        ctx[grow * DMODEL + col] = (bf16)(acc[c][r] * red[32 + rowi]);
      }
    }
  }
}

// ---------------------------------------------------------------------------
// Fused residual add + LayerNorm (D=512, block=256, one row per block).
// ---------------------------------------------------------------------------
__global__ __launch_bounds__(256)
void add_ln_kernel(const float* __restrict__ x, const float* __restrict__ y,
                   const float* __restrict__ g, const float* __restrict__ be,
                   float* __restrict__ outf, bf16* __restrict__ outb) {
  __shared__ float sred[16];
  const int row = blockIdx.x, tid = threadIdx.x;
  const float* xr = x + (size_t)row * DMODEL;
  const float* yr = y + (size_t)row * DMODEL;
  const float a0 = xr[tid] + yr[tid];
  const float a1 = xr[tid + 256] + yr[tid + 256];

  float s = a0 + a1;
  #pragma unroll
  for (int o = 16; o > 0; o >>= 1) s += __shfl_xor(s, o, 32);
  if ((tid & 31) == 0) sred[tid >> 5] = s;
  __syncthreads();
  if (tid < 32) {
    float t = (tid < 8) ? sred[tid] : 0.f;
    #pragma unroll
    for (int o = 4; o > 0; o >>= 1) t += __shfl_xor(t, o, 32);
    if (tid == 0) sred[8] = t;
  }
  __syncthreads();
  const float mean = sred[8] * (1.0f / DMODEL);
  const float d0 = a0 - mean, d1 = a1 - mean;

  float vs = d0 * d0 + d1 * d1;
  #pragma unroll
  for (int o = 16; o > 0; o >>= 1) vs += __shfl_xor(vs, o, 32);
  if ((tid & 31) == 0) sred[tid >> 5] = vs;
  __syncthreads();
  if (tid < 32) {
    float t = (tid < 8) ? sred[tid] : 0.f;
    #pragma unroll
    for (int o = 4; o > 0; o >>= 1) t += __shfl_xor(t, o, 32);
    if (tid == 0) sred[9] = t;
  }
  __syncthreads();
  const float inv = rsqrtf(sred[9] * (1.0f / DMODEL) + 1e-5f);

  const float o0 = d0 * inv * g[tid] + be[tid];
  const float o1 = d1 * inv * g[tid + 256] + be[tid + 256];
  float* of = outf + (size_t)row * DMODEL;
  of[tid] = o0;
  of[tid + 256] = o1;
  if (outb != nullptr) {
    outb[(size_t)row * DMODEL + tid]       = (bf16)o0;
    outb[(size_t)row * DMODEL + tid + 256] = (bf16)o1;
  }
}

// ---------------------------------------------------------------------------
// Launch
// ---------------------------------------------------------------------------
extern "C" void kernel_launch(void* const* d_in, const int* in_sizes, int n_in,
                              void* d_out, int out_size, void* d_ws, size_t ws_size,
                              hipStream_t stream) {
  const float* h_n     = (const float*)d_in[0];
  const float* to_v_b  = (const float*)d_in[2];
  const float* to_qk_b = (const float*)d_in[4];
  const float* proj_b  = (const float*)d_in[6];
  const float* ffn_b1  = (const float*)d_in[8];
  const float* ffn_b2  = (const float*)d_in[10];
  const float* ln1_g   = (const float*)d_in[11];
  const float* ln1_b   = (const float*)d_in[12];
  const float* ln2_g   = (const float*)d_in[13];
  const float* ln2_b   = (const float*)d_in[14];
  const long long* cum = (const long long*)d_in[15];

  const int T = in_sizes[0] / DMODEL;
  const int B = in_sizes[15] - 1;

  char* p = (char*)d_ws;
  auto carve = [&](size_t bytes) -> char* {
    char* r = p;
    p += (bytes + 255) & ~(size_t)255;
    return r;
  };
  bf16*  hnb  = (bf16*)carve((size_t)T * DMODEL * 2);
  bf16*  wvb  = (bf16*)carve((size_t)DMODEL * DMODEL * 2);
  bf16*  wqkb = (bf16*)carve((size_t)DMODEL * 2 * DMODEL * 2);
  bf16*  wpb  = (bf16*)carve((size_t)DMODEL * DMODEL * 2);
  bf16*  w1b  = (bf16*)carve((size_t)DMODEL * DFF * 2);
  bf16*  w2b  = (bf16*)carve((size_t)DFF * DMODEL * 2);
  bf16*  qb   = (bf16*)carve((size_t)T * DMODEL * 2);
  bf16*  kb   = (bf16*)carve((size_t)T * DMODEL * 2);
  bf16*  vb   = (bf16*)carve((size_t)T * DMODEL * 2);
  bf16*  ctxb = (bf16*)carve((size_t)T * DMODEL * 2);
  bf16*  h1b  = (bf16*)carve((size_t)T * DMODEL * 2);
  bf16*  midb = (bf16*)carve((size_t)T * DFF * 2);
  float* tmpf = (float*)carve((size_t)T * DMODEL * 4);
  float* h1f  = (float*)carve((size_t)T * DMODEL * 4);

  auto cvt = [&](const float* s, bf16* d, size_t n) {
    const int blocks = (int)(((n + 3) / 4 + 255) / 256);
    cvt_f32_to_bf16_kernel<<<blocks, 256, 0, stream>>>(s, d, (int)n);
  };
  cvt(h_n, hnb, (size_t)T * DMODEL);
  cvt((const float*)d_in[1], wvb,  (size_t)DMODEL * DMODEL);
  cvt((const float*)d_in[3], wqkb, (size_t)DMODEL * 2 * DMODEL);
  cvt((const float*)d_in[5], wpb,  (size_t)DMODEL * DMODEL);
  cvt((const float*)d_in[7], w1b,  (size_t)DMODEL * DFF);
  cvt((const float*)d_in[9], w2b,  (size_t)DFF * DMODEL);

  const int MB = (T + BM - 1) / BM;
  const dim3 blk(128);

  // v / q / k projections (bf16 out for attention)
  gemm_bf16_kernel<1><<<dim3(DMODEL / BN, MB), blk, 0, stream>>>(
      hnb, DMODEL, wvb, DMODEL, to_v_b, nullptr, vb, DMODEL, T, DMODEL);
  gemm_bf16_kernel<1><<<dim3(DMODEL / BN, MB), blk, 0, stream>>>(
      hnb, DMODEL, wqkb, 2 * DMODEL, to_qk_b, nullptr, qb, DMODEL, T, DMODEL);
  gemm_bf16_kernel<1><<<dim3(DMODEL / BN, MB), blk, 0, stream>>>(
      hnb, DMODEL, wqkb + DMODEL, 2 * DMODEL, to_qk_b + DMODEL, nullptr, kb, DMODEL, T, DMODEL);

  // attention
  attn_kernel<<<dim3(MAXN / 16, HEADS, B), dim3(32), ATTN_SMEM, stream>>>(
      qb, kb, vb, cum, ctxb);

  // output projection + residual LN1
  gemm_bf16_kernel<0><<<dim3(DMODEL / BN, MB), blk, 0, stream>>>(
      ctxb, DMODEL, wpb, DMODEL, proj_b, tmpf, nullptr, DMODEL, T, DMODEL);
  add_ln_kernel<<<T, 256, 0, stream>>>(h_n, tmpf, ln1_g, ln1_b, h1f, h1b);

  // FFN
  gemm_bf16_kernel<2><<<dim3(DFF / BN, MB), blk, 0, stream>>>(
      h1b, DMODEL, w1b, DFF, ffn_b1, nullptr, midb, DFF, T, DMODEL);
  gemm_bf16_kernel<0><<<dim3(DMODEL / BN, MB), blk, 0, stream>>>(
      midb, DFF, w2b, DMODEL, ffn_b2, tmpf, nullptr, DMODEL, T, DFF);

  // residual LN2 -> final output (f32)
  add_ln_kernel<<<T, 256, 0, stream>>>(h1f, tmpf, ln2_g, ln2_b, (float*)d_out, nullptr);
}
